// AGCNN_IA_88802743812389
// MI455X (gfx1250) — matrix operand order repaired and checked by reading, not compiled, not run on hardware
//
#include <hip/hip_runtime.h>
#include <math.h>

typedef __attribute__((ext_vector_type(16))) _Float16 v16h;
typedef __attribute__((ext_vector_type(8)))  float    v8f;

// ---------------- WMMA fragment helpers (wave32, 16x16x32 f16) ----------------
// A-matrix 16x32 f16 layout (ISA 7.12.2): lanes 0-15 row M=lane hold K={0..7,16..23},
// lanes 16-31 (same M) hold K={8..15,24..31}; pairs per VGPR.
__device__ __forceinline__ v16h frag_A(const _Float16* p, int lane) {
    const int khalf = ((lane >> 4) << 3);   // 0 or 8
    v16h f;
#pragma unroll
    for (int v = 0; v < 8; ++v) {
        const int kk = ((v < 4) ? (2 * v) : (2 * v + 8)) + khalf;
        f[2 * v]     = p[kk];
        f[2 * v + 1] = p[kk + 1];
    }
    return f;
}
// B-matrix 32x16 f16 layout: lanes 0-15 col N=lane hold K={0..7,16..23},
// lanes 16-31 hold K={8..15,24..31}. p points at row n of an NxK (B^T row-major) buffer.
__device__ __forceinline__ v16h frag_B(const _Float16* p, int lane) {
    const int khalf = ((lane >> 4) << 3);   // 0 or 8
    v16h f;
#pragma unroll
    for (int v = 0; v < 8; ++v) {
        const int kk = ((v < 4) ? (2 * v) : (2 * v + 8)) + khalf;
        f[2 * v]     = p[kk];
        f[2 * v + 1] = p[kk + 1];
    }
    return f;
}

#define WMMA_F16(Afr, Bfr, Cacc) \
    __builtin_amdgcn_wmma_f32_16x16x32_f16(false, (Afr), false, (Bfr), (short)0, (Cacc), false, false)

// ---------------- packing / norm kernels ----------------
__global__ void k_packT16(const float* __restrict__ F, _Float16* __restrict__ T,
                          int D, int N, int ldk, int total) {
    int i = blockIdx.x * blockDim.x + threadIdx.x;
    if (i >= total) return;
    int c = i % ldk;
    int bn = i / ldk;
    int n = bn % N, b = bn / N;
    T[i] = (c < D) ? (_Float16)F[((size_t)b * D + c) * N + n] : (_Float16)0.0f;
}

__global__ void k_norms(const float* __restrict__ F, float* __restrict__ xx,
                        int D, int N, int total) {
    int i = blockIdx.x * blockDim.x + threadIdx.x;
    if (i >= total) return;
    int b = i / N, n = i % N;
    float s = 0.f;
    for (int d = 0; d < D; ++d) {
        float v = F[((size_t)b * D + d) * N + n];
        s += v * v;
    }
    xx[i] = s;
}

__global__ void k_normT16(const float* __restrict__ F, _Float16* __restrict__ T,
                          int D, int N, int total) {
    int i = blockIdx.x * blockDim.x + threadIdx.x;
    if (i >= total) return;
    int b = i / N, n = i % N;
    float s = 0.f;
    for (int d = 0; d < D; ++d) {
        float v = F[((size_t)b * D + d) * N + n];
        s += v * v;
    }
    float inv = 1.0f / fmaxf(sqrtf(s), 1e-12f);
    _Float16* __restrict__ o = T + ((size_t)b * N + n) * D;
    for (int d = 0; d < D; ++d)
        o[d] = (_Float16)(F[((size_t)b * D + d) * N + n] * inv);
}

__global__ void k_packW(const float* __restrict__ W, _Float16* __restrict__ W16,
                        int O, int C, int Cp) {
    int i = blockIdx.x * blockDim.x + threadIdx.x;
    if (i >= O * Cp) return;
    int o = i / Cp, c = i % Cp;
    W16[i] = (c < C) ? (_Float16)W[o * C + c] : (_Float16)0.0f;
}

// ---------------- WMMA similarity, 2x2 register-blocked (32x32 per wave) ----------------
// S[b][i][j] = sum_k A[b][i][k]*Bm[b][j][k]; if xx!=null write 2*acc - xx[i] - xx[j].
__global__ void k_wmma_sim(const _Float16* __restrict__ A16, const _Float16* __restrict__ B16,
                           int lda, int ldb, int Kp, const float* __restrict__ xx,
                           float* __restrict__ S, int N) {
    const int lane = threadIdx.x;
    const int b = blockIdx.z;
    const int i0 = blockIdx.y * 32, j0 = blockIdx.x * 32;
    const int r = lane & 15;
    const _Float16* __restrict__ Arow0 = A16 + ((size_t)b * N + i0 + r) * lda;
    const _Float16* __restrict__ Arow1 = Arow0 + (size_t)16 * lda;
    const _Float16* __restrict__ Brow0 = B16 + ((size_t)b * N + j0 + r) * ldb;
    const _Float16* __restrict__ Brow1 = Brow0 + (size_t)16 * ldb;
    v8f acc00 = {}, acc01 = {}, acc10 = {}, acc11 = {};
    for (int k0 = 0; k0 < Kp; k0 += 32) {
        v16h a0 = frag_A(Arow0 + k0, lane);
        v16h a1 = frag_A(Arow1 + k0, lane);
        v16h b0 = frag_B(Brow0 + k0, lane);
        v16h b1 = frag_B(Brow1 + k0, lane);
        acc00 = WMMA_F16(a0, b0, acc00);
        acc01 = WMMA_F16(a0, b1, acc01);
        acc10 = WMMA_F16(a1, b0, acc10);
        acc11 = WMMA_F16(a1, b1, acc11);
    }
    const int nl = lane & 15;
    const int mb = ((lane >> 4) << 3);
    float* __restrict__ Sb = S + (size_t)b * N * N;
    const float* __restrict__ xb = xx ? (xx + (size_t)b * N) : nullptr;
    v8f accs[4] = {acc00, acc01, acc10, acc11};
#pragma unroll
    for (int t = 0; t < 4; ++t) {
        const int ti = t >> 1, tj = t & 1;
        const int m0 = i0 + 16 * ti + mb;
        const int n  = j0 + 16 * tj + nl;
        if (xb) {
            const float xn = xb[n];
#pragma unroll
            for (int g = 0; g < 8; ++g) {
                const int m = m0 + g;
                Sb[(size_t)m * N + n] = 2.0f * accs[t][g] - xb[m] - xn;
            }
        } else {
#pragma unroll
            for (int g = 0; g < 8; ++g)
                Sb[(size_t)(m0 + g) * N + n] = accs[t][g];
        }
    }
}

// ---------------- WMMA conv GEMM, 2x2 register-blocked ----------------
// Y[b][o][n] = sum_c W16[o][c] * BT[b][n][c]
__global__ void k_wmma_gemm(const _Float16* __restrict__ W16, const _Float16* __restrict__ BT,
                            float* __restrict__ Y, int O, int N, int Cp, int ldb) {
    const int lane = threadIdx.x;
    const int b = blockIdx.z;
    const int n0 = blockIdx.x * 32, o0 = blockIdx.y * 32;
    const int r = lane & 15;
    const _Float16* __restrict__ Arow0 = W16 + (size_t)(o0 + r) * Cp;
    const _Float16* __restrict__ Arow1 = Arow0 + (size_t)16 * Cp;
    const _Float16* __restrict__ Brow0 = BT + ((size_t)b * N + n0 + r) * ldb;
    const _Float16* __restrict__ Brow1 = Brow0 + (size_t)16 * ldb;
    v8f acc00 = {}, acc01 = {}, acc10 = {}, acc11 = {};
    for (int k0 = 0; k0 < Cp; k0 += 32) {
        v16h a0 = frag_A(Arow0 + k0, lane);
        v16h a1 = frag_A(Arow1 + k0, lane);
        v16h b0 = frag_B(Brow0 + k0, lane);
        v16h b1 = frag_B(Brow1 + k0, lane);
        acc00 = WMMA_F16(a0, b0, acc00);
        acc01 = WMMA_F16(a0, b1, acc01);
        acc10 = WMMA_F16(a1, b0, acc10);
        acc11 = WMMA_F16(a1, b1, acc11);
    }
    const int nl = lane & 15;
    const int ob = ((lane >> 4) << 3);
    float* __restrict__ Yb = Y + (size_t)b * O * N;
    v8f accs[4] = {acc00, acc01, acc10, acc11};
#pragma unroll
    for (int t = 0; t < 4; ++t) {
        const int ti = t >> 1, tj = t & 1;
        const int o = o0 + 16 * ti + ob;
        const int n = n0 + 16 * tj + nl;
#pragma unroll
        for (int g = 0; g < 8; ++g)
            Yb[(size_t)(o + g) * N + n] = accs[t][g];
    }
}

// ---------------- top-k (k=20) + softmax over selected values ----------------
__global__ void k_topk_softmax(const float* __restrict__ S, int* __restrict__ Idx,
                               float* __restrict__ Wt, int N, int K) {
    __shared__ float row[2048];
    __shared__ float rv[256];
    __shared__ int   ri[256];
    __shared__ float selv[32];
    __shared__ int   seli[32];
    const size_t base = (size_t)blockIdx.x * N;   // blockIdx.x = b*N + i
    const int t = threadIdx.x;
    for (int j = t; j < N; j += 256) row[j] = S[base + j];
    __syncthreads();
    for (int k = 0; k < K; ++k) {
        float bv = -__builtin_inff();
        int   bi = N;
        for (int j = t; j < N; j += 256) {
            float v = row[j];
            if (v > bv || (v == bv && j < bi)) { bv = v; bi = j; }
        }
        rv[t] = bv; ri[t] = bi;
        __syncthreads();
        for (int s = 128; s > 0; s >>= 1) {
            if (t < s) {
                if (rv[t + s] > rv[t] || (rv[t + s] == rv[t] && ri[t + s] < ri[t])) {
                    rv[t] = rv[t + s]; ri[t] = ri[t + s];
                }
            }
            __syncthreads();
        }
        if (t == 0) {
            selv[k] = rv[0]; seli[k] = ri[0];
            row[ri[0]] = -__builtin_inff();
        }
        __syncthreads();
    }
    if (t == 0) {
        float m = selv[0];
        float e[32];
        float sum = 0.f;
        for (int k = 0; k < K; ++k) { e[k] = expf(selv[k] - m); sum += e[k]; }
        float inv = 1.0f / sum;
        for (int k = 0; k < K; ++k) {
            Idx[(size_t)blockIdx.x * K + k] = seli[k];
            Wt [(size_t)blockIdx.x * K + k] = e[k] * inv;
        }
    }
}

// ---------------- attention aggregation -> AttT16 [B][N][Cp] ----------------
__global__ void k_agg(const float* __restrict__ Fdst, const float* __restrict__ Fsrc,
                      const int* __restrict__ Idx, const float* __restrict__ Wt,
                      _Float16* __restrict__ AttT, int D, int N, int Cp, int K) {
    const int n = blockIdx.x, b = blockIdx.y, c = threadIdx.x;
    __shared__ int   sI[32];
    __shared__ float sW[32];
    if (c < K) {
        sI[c] = Idx[((size_t)b * N + n) * K + c];
        sW[c] = Wt [((size_t)b * N + n) * K + c];
    }
    __syncthreads();
    _Float16 outv = (_Float16)0.0f;
    if (c < D) {
        outv = (_Float16)Fdst[((size_t)b * D + c) * N + n];
    } else if (c < 2 * D) {
        const int d = c - D;
        const float* __restrict__ src = Fsrc + ((size_t)b * D + d) * N;
        float s = 0.f;
        for (int k = 0; k < K; ++k) s += sW[k] * src[sI[k]];
        s -= Fdst[((size_t)b * D + d) * N + n];
        outv = (_Float16)s;
    }
    AttT[((size_t)b * N + n) * Cp + c] = outv;
}

// ---------------- BatchNorm (training-mode batch stats) ----------------
__global__ void k_bn_stats(const float* __restrict__ Y, float* __restrict__ mean,
                           float* __restrict__ istd, int O, int N, int B) {
    const int o = blockIdx.x, t = threadIdx.x;
    float s = 0.f, s2 = 0.f;
    for (int b = 0; b < B; ++b) {
        const float* __restrict__ p = Y + ((size_t)b * O + o) * N;
        for (int n = t; n < N; n += 256) { float v = p[n]; s += v; s2 += v * v; }
    }
    __shared__ float sh[256], sh2[256];
    sh[t] = s; sh2[t] = s2;
    __syncthreads();
    for (int st = 128; st > 0; st >>= 1) {
        if (t < st) { sh[t] += sh[t + st]; sh2[t] += sh2[t + st]; }
        __syncthreads();
    }
    if (t == 0) {
        float cnt = (float)(B * N);
        float m = sh[0] / cnt;
        float v = sh2[0] / cnt - m * m;
        mean[o] = m;
        istd[o] = rsqrtf(fmaxf(v, 0.f) + 1e-5f);
    }
}

__global__ void k_bn_apply(const float* __restrict__ Y, const float* __restrict__ mean,
                           const float* __restrict__ istd, const float* __restrict__ g,
                           const float* __restrict__ beta, float* __restrict__ Xout,
                           _Float16* __restrict__ T16, int ldT, int coff,
                           int O, int N) {
    const size_t i = (size_t)blockIdx.x * 256 + threadIdx.x;   // o*N + n
    if (i >= (size_t)O * N) return;
    const int b = blockIdx.y;
    const int o = (int)(i / N), n = (int)(i % N);
    float y = Y[((size_t)b * O + o) * N + n];
    float yh = (y - mean[o]) * istd[o] * g[o] + beta[o];
    float r = (yh >= 0.f) ? yh : 0.01f * yh;
    if (Xout) Xout[((size_t)b * O + o) * N + n] = r;
    if (T16)  T16[((size_t)b * N + n) * ldT + coff + o] = (_Float16)r;
}

// ---------------- host launch ----------------
static inline size_t align256(size_t x) { return (x + 255) & ~(size_t)255; }

extern "C" void kernel_launch(void* const* d_in, const int* in_sizes, int n_in,
                              void* d_out, int out_size, void* d_ws, size_t ws_size,
                              hipStream_t stream) {
    (void)in_sizes; (void)n_in; (void)out_size; (void)ws_size;
    const int B = 8, N = 2048, K = 20;

    const float* x  = (const float*)d_in[0];   // [8,3,2048]
    const float* y3 = (const float*)d_in[1];   // [8,128,2048]
    const float* w1 = (const float*)d_in[2];   // [64,6]
    const float* w2 = (const float*)d_in[3];   // [64,128]
    const float* w3 = (const float*)d_in[4];   // [128,128]
    const float* w4 = (const float*)d_in[5];   // [256,256]
    const float* w5 = (const float*)d_in[6];   // [512,512]
    const float* g1 = (const float*)d_in[7],  *b1 = (const float*)d_in[8];
    const float* g2 = (const float*)d_in[9],  *b2 = (const float*)d_in[10];
    const float* g3 = (const float*)d_in[11], *b3 = (const float*)d_in[12];
    const float* g4 = (const float*)d_in[13], *b4 = (const float*)d_in[14];
    const float* g5 = (const float*)d_in[15], *b5 = (const float*)d_in[16];
    float* out = (float*)d_out;                // [8,512,2048]

    // ---- workspace carve ----
    char* w = (char*)d_ws;
    size_t off = 0;
    auto take = [&](size_t bytes) -> void* { void* p = w + off; off += align256(bytes); return p; };
    float*     dist  = (float*)    take((size_t)B * N * N * 4);     // 134 MB
    int*       Idx   = (int*)      take((size_t)B * N * K * 4);
    float*     Wt    = (float*)    take((size_t)B * N * K * 4);
    _Float16*  XcT   = (_Float16*) take((size_t)B * N * 512 * 2);   // concat f16; stage-2/3 sim operand
    _Float16*  AttT  = (_Float16*) take((size_t)B * N * 256 * 2);   // max Cp = 256
    _Float16*  Xt16a = (_Float16*) take((size_t)B * N * 32 * 2);
    _Float16*  XnT   = (_Float16*) take((size_t)B * N * 128 * 2);
    _Float16*  YnT   = (_Float16*) take((size_t)B * N * 128 * 2);
    float*     xx    = (float*)    take((size_t)B * N * 4);
    float*     Yf    = (float*)    take((size_t)B * 512 * N * 4);
    float*     x1f   = (float*)    take((size_t)B * 64  * N * 4);
    float*     x2f   = (float*)    take((size_t)B * 64  * N * 4);
    float*     x3f   = (float*)    take((size_t)B * 128 * N * 4);
    _Float16*  W16a  = (_Float16*) take((size_t)64  * 32  * 2);
    _Float16*  W16b  = (_Float16*) take((size_t)64  * 128 * 2);
    _Float16*  W16c  = (_Float16*) take((size_t)128 * 128 * 2);
    _Float16*  W16d  = (_Float16*) take((size_t)256 * 256 * 2);
    _Float16*  W16e  = (_Float16*) take((size_t)512 * 512 * 2);
    float*     mean  = (float*)    take(512 * 4);
    float*     istd  = (float*)    take(512 * 4);

    const int T32 = N / 32;                 // 64 macro-tiles
    const dim3 simGrid(T32, T32, B);
    const dim3 wave(32);

    // ---- weight packing ----
    k_packW<<<(64 * 32 + 255) / 256, 256, 0, stream>>>(w1, W16a, 64, 6, 32);
    k_packW<<<(64 * 128 + 255) / 256, 256, 0, stream>>>(w2, W16b, 64, 128, 128);
    k_packW<<<(128 * 128 + 255) / 256, 256, 0, stream>>>(w3, W16c, 128, 128, 128);
    k_packW<<<(256 * 256 + 255) / 256, 256, 0, stream>>>(w4, W16d, 256, 256, 256);
    k_packW<<<(512 * 512 + 255) / 256, 256, 0, stream>>>(w5, W16e, 512, 512, 512);

    // ================= stage 1: self-att(x, D=3) -> conv(6->64) =================
    k_packT16<<<(B * N * 32 + 255) / 256, 256, 0, stream>>>(x, Xt16a, 3, N, 32, B * N * 32);
    k_norms<<<(B * N + 255) / 256, 256, 0, stream>>>(x, xx, 3, N, B * N);
    k_wmma_sim<<<simGrid, wave, 0, stream>>>(Xt16a, Xt16a, 32, 32, 32, xx, dist, N);
    k_topk_softmax<<<B * N, 256, 0, stream>>>(dist, Idx, Wt, N, K);
    k_agg<<<dim3(N, B), 32, 0, stream>>>(x, x, Idx, Wt, AttT, 3, N, 32, K);
    k_wmma_gemm<<<dim3(T32, 64 / 32, B), wave, 0, stream>>>(W16a, AttT, Yf, 64, N, 32, 32);
    k_bn_stats<<<64, 256, 0, stream>>>(Yf, mean, istd, 64, N, B);
    k_bn_apply<<<dim3(64 * N / 256, B), 256, 0, stream>>>(Yf, mean, istd, g1, b1,
                                                          x1f, XcT, 512, 0, 64, N);

    // ================= stage 2: self-att(x1, D=64) -> conv(128->64) =================
    k_norms<<<(B * N + 255) / 256, 256, 0, stream>>>(x1f, xx, 64, N, B * N);
    k_wmma_sim<<<simGrid, wave, 0, stream>>>(XcT, XcT, 512, 512, 64, xx, dist, N);
    k_topk_softmax<<<B * N, 256, 0, stream>>>(dist, Idx, Wt, N, K);
    k_agg<<<dim3(N, B), 128, 0, stream>>>(x1f, x1f, Idx, Wt, AttT, 64, N, 128, K);
    k_wmma_gemm<<<dim3(T32, 64 / 32, B), wave, 0, stream>>>(W16b, AttT, Yf, 64, N, 128, 128);
    k_bn_stats<<<64, 256, 0, stream>>>(Yf, mean, istd, 64, N, B);
    k_bn_apply<<<dim3(64 * N / 256, B), 256, 0, stream>>>(Yf, mean, istd, g2, b2,
                                                          x2f, XcT, 512, 64, 64, N);

    // ================= stage 3: self-att(x2, D=64) -> conv(128->128) =================
    k_norms<<<(B * N + 255) / 256, 256, 0, stream>>>(x2f, xx, 64, N, B * N);
    k_wmma_sim<<<simGrid, wave, 0, stream>>>(XcT + 64, XcT + 64, 512, 512, 64, xx, dist, N);
    k_topk_softmax<<<B * N, 256, 0, stream>>>(dist, Idx, Wt, N, K);
    k_agg<<<dim3(N, B), 128, 0, stream>>>(x2f, x2f, Idx, Wt, AttT, 64, N, 128, K);
    k_wmma_gemm<<<dim3(T32, 128 / 32, B), wave, 0, stream>>>(W16c, AttT, Yf, 128, N, 128, 128);
    k_bn_stats<<<128, 256, 0, stream>>>(Yf, mean, istd, 128, N, B);
    k_bn_apply<<<dim3(128 * N / 256, B), 256, 0, stream>>>(Yf, mean, istd, g3, b3,
                                                           x3f, XcT, 512, 128, 128, N);

    // ================= stage 4: cross-att(x3, y3, D=128) -> conv(256->256) =================
    k_normT16<<<(B * N + 255) / 256, 256, 0, stream>>>(x3f, XnT, 128, N, B * N);
    k_normT16<<<(B * N + 255) / 256, 256, 0, stream>>>(y3, YnT, 128, N, B * N);
    k_wmma_sim<<<simGrid, wave, 0, stream>>>(XnT, YnT, 128, 128, 128, nullptr, dist, N);
    k_topk_softmax<<<B * N, 256, 0, stream>>>(dist, Idx, Wt, N, K);
    k_agg<<<dim3(N, B), 256, 0, stream>>>(x3f, y3, Idx, Wt, AttT, 128, N, 256, K);
    k_wmma_gemm<<<dim3(T32, 256 / 32, B), wave, 0, stream>>>(W16d, AttT, Yf, 256, N, 256, 256);
    k_bn_stats<<<256, 256, 0, stream>>>(Yf, mean, istd, 256, N, B);
    k_bn_apply<<<dim3(256 * N / 256, B), 256, 0, stream>>>(Yf, mean, istd, g4, b4,
                                                           nullptr, XcT, 512, 256, 256, N);

    // ================= stage 5: conv(512->512) on concat =================
    k_wmma_gemm<<<dim3(T32, 512 / 32, B), wave, 0, stream>>>(W16e, XcT, Yf, 512, N, 512, 512);
    k_bn_stats<<<512, 256, 0, stream>>>(Yf, mean, istd, 512, N, B);
    k_bn_apply<<<dim3(512 * N / 256, B), 256, 0, stream>>>(Yf, mean, istd, g5, b5,
                                                           out, nullptr, 0, 0, 512, N);
}
